// GCNCritic_13606456394316
// MI455X (gfx1250) — compile-verified, matrix operand order checked by reading, not run on hardware
//
#include <hip/hip_runtime.h>
#include <hip/hip_bf16.h>
#include <stdint.h>

// ---------------------------------------------------------------------------
// Types for CDNA5 WMMA (wave32): D(16x16 f32) = A(16x32 bf16) * B(32x16 bf16) + C
// ---------------------------------------------------------------------------
typedef __attribute__((ext_vector_type(16))) __bf16    v16bf;
typedef __attribute__((ext_vector_type(8)))  float     v8f;
typedef __attribute__((ext_vector_type(8)))  unsigned short v8us;

#define OBSD 128
#define NAGT 16
#define NACT 8
#define HID  128
#define GED  256
#define LED  256
#define BATCH 2048
#define NNODE (BATCH * NAGT)   // 32768

// Workspace layout (units: uint16 / bf16 elements). All offsets are 1KB-multiples.
#define OFF_WPRE  ((size_t)0)
#define OFF_WLOC  (OFF_WPRE  + (size_t)128*128)
#define OFF_WG0   (OFF_WLOC  + (size_t)128*256)
#define OFF_WG1   (OFF_WG0   + (size_t)128*128)
#define OFF_WPOST (OFF_WG1   + (size_t)128*128)
#define OFF_W1    (OFF_WPOST + (size_t)128*256)
#define OFF_W2    (OFF_W1    + (size_t)512*512)
#define OFF_W3    (OFF_W2    + (size_t)512*512)
#define OFF_X0    (OFF_W3    + (size_t)512*16)
#define OFF_LOC   (OFF_X0    + (size_t)NNODE*HID)
#define OFF_G     (OFF_LOC   + (size_t)NNODE*LED)

// ---------------------------------------------------------------------------
// bf16 helpers (round-to-nearest-even via bit ops)
// ---------------------------------------------------------------------------
__device__ __forceinline__ unsigned short bf16bits(float f) {
  union { float f; unsigned int u; } c; c.f = f;
  unsigned int u = c.u;
  unsigned int r = u + 0x7FFFu + ((u >> 16) & 1u);
  return (unsigned short)(r >> 16);
}
__device__ __forceinline__ float bf16tof(unsigned short b) {
  union { unsigned int u; float f; } c; c.u = ((unsigned int)b) << 16;
  return c.f;
}

// ---------------------------------------------------------------------------
// Fragment loaders following ISA 7.12.2 layouts (wave32).
// A (16x32 bf16): lane L -> row M=L&15; half=L>>4;
//   elements 0..7  : K = kb + 8*half + {0..7}
//   elements 8..15 : K = kb + 16 + 8*half + {0..7}
// B (32x16 bf16): pre-swizzled -> one contiguous 32B load per lane.
// ---------------------------------------------------------------------------
union AFragU { v16bf v; v8us h[2]; };

__device__ __forceinline__ v16bf load_A_lds(const unsigned short* lds, int stride,
                                            int kb, int lane) {
  const int m = lane & 15, half = lane >> 4;
  const unsigned short* p = lds + (size_t)m * stride + kb + half * 8;
  AFragU u;
  u.h[0] = *(const v8us*)(p);
  u.h[1] = *(const v8us*)(p + 16);
  return u.v;
}

__device__ __forceinline__ v16bf load_A_global_f32(const float* base, int stride,
                                                   int kb, int lane) {
  const int m = lane & 15, half = lane >> 4;
  const float* p0 = base + (size_t)m * stride + kb + half * 8;
  const float* p1 = p0 + 16;
  v16bf a;
#pragma unroll
  for (int i = 0; i < 8; ++i) {
    a[i]     = __builtin_bit_cast(__bf16, bf16bits(p0[i]));
    a[i + 8] = __builtin_bit_cast(__bf16, bf16bits(p1[i]));
  }
  return a;
}

__device__ __forceinline__ v16bf load_B_packed(const unsigned short* pw, int frag,
                                               int lane) {
  return *(const v16bf*)(pw + (size_t)frag * 512 + lane * 16);
}

__device__ __forceinline__ v8f wmma_bf16(v16bf a, v16bf b, v8f c) {
  return __builtin_amdgcn_wmma_f32_16x16x32_bf16(false, a, false, b, (short)0, c,
                                                 false, false);
}

// D layout: lane L -> col N=L&15; half=L>>4; VGPR r -> row M=r+8*half.
__device__ __forceinline__ void store_D_lds(unsigned short* lds, int stride, int nb,
                                            int lane, v8f d, float bn, bool relu) {
  const int n = lane & 15, half = lane >> 4;
#pragma unroll
  for (int r = 0; r < 8; ++r) {
    float v = d[r] + bn;
    if (relu && v < 0.0f) v = 0.0f;
    lds[(size_t)(r + 8 * half) * stride + nb + n] = bf16bits(v);
  }
}

__device__ __forceinline__ void store_D_global(unsigned short* g, int stride,
                                               size_t rowbase, int nb, int lane,
                                               v8f d, float bn, bool relu) {
  const int n = lane & 15, half = lane >> 4;
#pragma unroll
  for (int r = 0; r < 8; ++r) {
    float v = d[r] + bn;
    if (relu && v < 0.0f) v = 0.0f;
    g[(rowbase + r + 8 * half) * (size_t)stride + nb + n] = bf16bits(v);
  }
}

// ---------------------------------------------------------------------------
// K0: pack row-major f32 weight [K,N] into bf16 WMMA B fragments.
// idx: frag=idx>>9, lane=(idx>>4)&31, e=idx&15; frag=ct*(K/32)+kt;
// N=ct*16+(lane&15); K=kt*32 + (e<8 ? 8*half+e : 16+8*half+(e-8)).
// ---------------------------------------------------------------------------
__global__ void k_pack_b(const float* __restrict__ W, unsigned short* __restrict__ dst,
                         int K, int N, int Npad) {
  const int idx = blockIdx.x * blockDim.x + threadIdx.x;
  const int total = K * Npad;
  if (idx >= total) return;
  const int f = idx >> 9;
  const int lane = (idx >> 4) & 31;
  const int e = idx & 15;
  const int ktc = K >> 5;
  const int ct = f / ktc;
  const int kt = f - ct * ktc;
  const int n = ct * 16 + (lane & 15);
  const int half = lane >> 4;
  const int kl = (e < 8) ? (8 * half + e) : (16 + 8 * half + (e - 8));
  const int k = kt * 32 + kl;
  const float v = (n < N) ? W[(size_t)k * N + n] : 0.0f;
  dst[idx] = bf16bits(v);
}

// ---------------------------------------------------------------------------
// K1: x0 = relu(obs@W_pre+b_pre) [N,128]; loc = relu(obs@W_loc+b_loc) [N,256].
// One block per 16-node tile; each wave loads the 4 shared A fragments ONCE,
// then runs 3 column tiles (1 of x0, 2 of loc) as multi-accumulator WMMAs.
// ---------------------------------------------------------------------------
__global__ __launch_bounds__(256) void k_pre_loc(
    const float* __restrict__ obs,
    const unsigned short* __restrict__ Wpre_p, const float* __restrict__ b_pre,
    const unsigned short* __restrict__ Wloc_p, const float* __restrict__ b_loc,
    unsigned short* __restrict__ x0, unsigned short* __restrict__ locb) {
  const int tile = blockIdx.x;                 // 0..2047
  const size_t rowbase = (size_t)tile * 16;
  const int wave = threadIdx.x >> 5;
  const int lane = threadIdx.x & 31;
  const float* obs_tile = obs + rowbase * OBSD;

  v16bf a[4];
#pragma unroll
  for (int kt = 0; kt < 4; ++kt)
    a[kt] = load_A_global_f32(obs_tile, OBSD, kt * 32, lane);

  v8f acc0 = {}, acc1 = {}, acc2 = {};
#pragma unroll
  for (int kt = 0; kt < 4; ++kt) {             // K = 128
    acc0 = wmma_bf16(a[kt], load_B_packed(Wpre_p, wave * 4 + kt, lane), acc0);
    acc1 = wmma_bf16(a[kt], load_B_packed(Wloc_p, wave * 4 + kt, lane), acc1);
    acc2 = wmma_bf16(a[kt], load_B_packed(Wloc_p, (wave + 8) * 4 + kt, lane), acc2);
  }
  store_D_global(x0, HID, rowbase, wave * 16, lane, acc0,
                 b_pre[wave * 16 + (lane & 15)], true);
  store_D_global(locb, LED, rowbase, wave * 16, lane, acc1,
                 b_loc[wave * 16 + (lane & 15)], true);
  store_D_global(locb, LED, rowbase, (wave + 8) * 16, lane, acc2,
                 b_loc[(wave + 8) * 16 + (lane & 15)], true);
}

// ---------------------------------------------------------------------------
// K2: collapsed GCN chain per 16-group tile:
//   m  = mean_agents(x0)            [16,128] (LDS, vectorized staging)
//   x1 = relu(m @Wg0 + b_gcn[0])    [16,128]
//   x2 = relu(x1@Wg1 + b_gcn[1])    [16,128] (reuses m buffer)
//   g  = relu(x2@W_post + b_post)   [16,256] -> global
// ---------------------------------------------------------------------------
__global__ __launch_bounds__(256) void k_graph(
    const unsigned short* __restrict__ x0,
    const unsigned short* __restrict__ Wg0_p,
    const unsigned short* __restrict__ Wg1_p, const float* __restrict__ b_gcn,
    const unsigned short* __restrict__ Wpost_p, const float* __restrict__ b_post,
    unsigned short* __restrict__ gbuf) {
  __shared__ __align__(32) unsigned short sm[16 * HID];   // m, then x2
  __shared__ __align__(32) unsigned short sx1[16 * HID];

  const int tile = blockIdx.x;                 // 0..127
  const size_t gbase = (size_t)tile * 16;
  const int tid = threadIdx.x;
  const int wave = tid >> 5;
  const int lane = tid & 31;

  // Stage m: 256 chunks of 8 bf16 (one per thread), 16B vector loads.
  {
    const int row = tid >> 4;                  // group within tile
    const int col = (tid & 15) * 8;
    const size_t node0 = (gbase + row) * NAGT;
    float s[8] = {0, 0, 0, 0, 0, 0, 0, 0};
#pragma unroll
    for (int a = 0; a < NAGT; ++a) {
      v8us v = *(const v8us*)(x0 + (node0 + a) * HID + col);
#pragma unroll
      for (int i = 0; i < 8; ++i) s[i] += bf16tof(v[i]);
    }
    v8us o;
#pragma unroll
    for (int i = 0; i < 8; ++i) o[i] = bf16bits(s[i] * (1.0f / 16.0f));
    *(v8us*)(sm + row * HID + col) = o;
  }
  __syncthreads();

  // x1 = relu(m @ Wg0 + b_gcn[0])  (8 column tiles, one per wave)
  {
    v8f acc = {};
#pragma unroll
    for (int kt = 0; kt < 4; ++kt)
      acc = wmma_bf16(load_A_lds(sm, HID, kt * 32, lane),
                      load_B_packed(Wg0_p, wave * 4 + kt, lane), acc);
    store_D_lds(sx1, HID, wave * 16, lane, acc, b_gcn[wave * 16 + (lane & 15)], true);
  }
  __syncthreads();

  // x2 = relu(x1 @ Wg1 + b_gcn[1])  -> reuse sm
  {
    v8f acc = {};
#pragma unroll
    for (int kt = 0; kt < 4; ++kt)
      acc = wmma_bf16(load_A_lds(sx1, HID, kt * 32, lane),
                      load_B_packed(Wg1_p, wave * 4 + kt, lane), acc);
    store_D_lds(sm, HID, wave * 16, lane, acc,
                b_gcn[HID + wave * 16 + (lane & 15)], true);
  }
  __syncthreads();

  // g = relu(x2 @ W_post + b_post): 16 column tiles, 2 per wave, shared A.
  {
    v8f acc0 = {}, acc1 = {};
#pragma unroll
    for (int kt = 0; kt < 4; ++kt) {
      v16bf a = load_A_lds(sm, HID, kt * 32, lane);
      acc0 = wmma_bf16(a, load_B_packed(Wpost_p, wave * 4 + kt, lane), acc0);
      acc1 = wmma_bf16(a, load_B_packed(Wpost_p, (wave + 8) * 4 + kt, lane), acc1);
    }
    store_D_global(gbuf, GED, gbase, wave * 16, lane, acc0,
                   b_post[wave * 16 + (lane & 15)], true);
    store_D_global(gbuf, GED, gbase, (wave + 8) * 16, lane, acc1,
                   b_post[(wave + 8) * 16 + (lane & 15)], true);
  }
}

// ---------------------------------------------------------------------------
// K3: fused final MLP per 16-node tile (one group):
//   z  = [g[group] broadcast | loc rows]  [16,512] (LDS, 16B vector staging)
//   z1 = relu(z  @ W1 + b1)               [16,512]
//   z2 = relu(z1 @ W2 + b2)               [16,512] (reuses z)
//   q  = z2 @ W3 + b3                     [16,8]  -> d_out f32
// K-outer with 4 accumulators: 1 LDS A-fragment load feeds 4 WMMAs.
// ---------------------------------------------------------------------------
__global__ __launch_bounds__(256) void k_mlp(
    const unsigned short* __restrict__ gbuf,
    const unsigned short* __restrict__ locb,
    const unsigned short* __restrict__ W1_p, const float* __restrict__ b1,
    const unsigned short* __restrict__ W2_p, const float* __restrict__ b2,
    const unsigned short* __restrict__ W3_p, const float* __restrict__ b3,
    float* __restrict__ out) {
  __shared__ __align__(32) unsigned short sz[16 * 512];   // z, then z2
  __shared__ __align__(32) unsigned short sz1[16 * 512];

  const int tile = blockIdx.x;                 // 0..2047 == group id
  const size_t rowbase = (size_t)tile * 16;
  const int tid = threadIdx.x;
  const int wave = tid >> 5;
  const int lane = tid & 31;

  // Stage z with 16B chunks: 1024 chunks, 4 per thread.
#pragma unroll
  for (int it = 0; it < 4; ++it) {
    const int chunk = tid + it * 256;
    const int row = chunk >> 6;                // 64 chunks per row
    const int col = (chunk & 63) * 8;
    const unsigned short* src =
        (col < GED) ? (gbuf + (size_t)tile * GED + col)
                    : (locb + (rowbase + row) * LED + (col - GED));
    *(v8us*)(sz + (size_t)chunk * 8) = *(const v8us*)src;
  }
  __syncthreads();

  // z1 = relu(z @ W1 + b1): 32 column tiles; wave handles {w, w+8, w+16, w+24}.
  {
    v8f acc[4] = {{}, {}, {}, {}};
    for (int kt = 0; kt < 16; ++kt) {
      v16bf a = load_A_lds(sz, 512, kt * 32, lane);
#pragma unroll
      for (int j = 0; j < 4; ++j)
        acc[j] = wmma_bf16(a, load_B_packed(W1_p, (wave + 8 * j) * 16 + kt, lane),
                           acc[j]);
    }
#pragma unroll
    for (int j = 0; j < 4; ++j) {
      const int ct = wave + 8 * j;
      store_D_lds(sz1, 512, ct * 16, lane, acc[j], b1[ct * 16 + (lane & 15)], true);
    }
  }
  __syncthreads();

  // z2 = relu(z1 @ W2 + b2) -> reuse sz.
  {
    v8f acc[4] = {{}, {}, {}, {}};
    for (int kt = 0; kt < 16; ++kt) {
      v16bf a = load_A_lds(sz1, 512, kt * 32, lane);
#pragma unroll
      for (int j = 0; j < 4; ++j)
        acc[j] = wmma_bf16(a, load_B_packed(W2_p, (wave + 8 * j) * 16 + kt, lane),
                           acc[j]);
    }
#pragma unroll
    for (int j = 0; j < 4; ++j) {
      const int ct = wave + 8 * j;
      store_D_lds(sz, 512, ct * 16, lane, acc[j], b2[ct * 16 + (lane & 15)], true);
    }
  }
  __syncthreads();

  // q = z2 @ W3 + b3 (N padded 8->16; only n<8 stored). Wave 0 only.
  if (wave == 0) {
    v8f acc = {};
    for (int kt = 0; kt < 16; ++kt)
      acc = wmma_bf16(load_A_lds(sz, 512, kt * 32, lane),
                      load_B_packed(W3_p, kt, lane), acc);
    const int n = lane & 15, half = lane >> 4;
    if (n < NACT) {
      const float bn = b3[n];
#pragma unroll
      for (int r = 0; r < 8; ++r)
        out[(rowbase + r + 8 * half) * NACT + n] = acc[r] + bn;
    }
  }
}

// ---------------------------------------------------------------------------
// Host-side launch (graph-capture safe: launches only, all on `stream`).
// Input order: obs_j, W_pre, b_pre, W_gcn, b_gcn, W_post, b_post,
//              W_loc, b_loc, W1, b1, W2, b2, W3, b3
// ---------------------------------------------------------------------------
extern "C" void kernel_launch(void* const* d_in, const int* in_sizes, int n_in,
                              void* d_out, int out_size, void* d_ws, size_t ws_size,
                              hipStream_t stream) {
  const float* obs    = (const float*)d_in[0];
  const float* W_pre  = (const float*)d_in[1];
  const float* b_pre  = (const float*)d_in[2];
  const float* W_gcn  = (const float*)d_in[3];   // [2,128,128]
  const float* b_gcn  = (const float*)d_in[4];   // [2,128]
  const float* W_post = (const float*)d_in[5];
  const float* b_post = (const float*)d_in[6];
  const float* W_loc  = (const float*)d_in[7];
  const float* b_loc  = (const float*)d_in[8];
  const float* W1     = (const float*)d_in[9];
  const float* b1     = (const float*)d_in[10];
  const float* W2     = (const float*)d_in[11];
  const float* b2     = (const float*)d_in[12];
  const float* W3     = (const float*)d_in[13];
  const float* b3     = (const float*)d_in[14];
  float* out = (float*)d_out;

  unsigned short* ws = (unsigned short*)d_ws;
  unsigned short* pWpre  = ws + OFF_WPRE;
  unsigned short* pWloc  = ws + OFF_WLOC;
  unsigned short* pWg0   = ws + OFF_WG0;
  unsigned short* pWg1   = ws + OFF_WG1;
  unsigned short* pWpost = ws + OFF_WPOST;
  unsigned short* pW1    = ws + OFF_W1;
  unsigned short* pW2    = ws + OFF_W2;
  unsigned short* pW3    = ws + OFF_W3;
  unsigned short* x0     = ws + OFF_X0;
  unsigned short* locb   = ws + OFF_LOC;
  unsigned short* gbuf   = ws + OFF_G;

  auto pack = [&](const float* W, unsigned short* dst, int K, int N, int Npad) {
    const int total = K * Npad;
    k_pack_b<<<(total + 255) / 256, 256, 0, stream>>>(W, dst, K, N, Npad);
  };
  pack(W_pre,             pWpre,  128, 128, 128);
  pack(W_loc,             pWloc,  128, 256, 256);
  pack(W_gcn,             pWg0,   128, 128, 128);
  pack(W_gcn + 128 * 128, pWg1,   128, 128, 128);
  pack(W_post,            pWpost, 128, 256, 256);
  pack(W1,                pW1,    512, 512, 512);
  pack(W2,                pW2,    512, 512, 512);
  pack(W3,                pW3,    512,   8,  16);

  k_pre_loc<<<BATCH, 256, 0, stream>>>(obs, pWpre, b_pre, pWloc, b_loc, x0, locb);
  k_graph<<<BATCH / 16, 256, 0, stream>>>(x0, pWg0, pWg1, b_gcn, pWpost, b_post, gbuf);
  k_mlp<<<BATCH, 256, 0, stream>>>(gbuf, locb, pW1, b1, pW2, b2, pW3, b3, out);
}
